// SimpleResRNN_62543313764707
// MI455X (gfx1250) — compile-verified, matrix operand order checked by reading, not run on hardware
//
#include <hip/hip_runtime.h>

#define DT_C    0.01f
#define THETA_C 1.0f

#define T_STEPS 250
#define B_SZ    256
#define IN_SZ   700
#define H_SZ    256
#define OUT_SZ  20
#define M_ROWS  (T_STEPS * B_SZ)   // 64000
#define K_PAD   704

// ---- d_out layout (floats): outs, fz, fu, fout, spike_sum
#define OUTS_OFF 0
#define FZ_OFF   (T_STEPS * B_SZ * OUT_SZ)      // 1,280,000
#define FU_OFF   (FZ_OFF + B_SZ * H_SZ)         // 1,345,536
#define FOUT_OFF (FU_OFF + B_SZ * H_SZ)         // 1,411,072
#define SS_OFF   (FOUT_OFF + B_SZ * OUT_SZ)     // 1,416,192

// ---- workspace layout (floats)
#define WS_OM   0
#define WS_PB   256
#define WS_AL   512
#define WS_CURX 1024   // permuted cur_x: [(m>>4)][h][m&15] = 16,384,000 floats

typedef __attribute__((ext_vector_type(16))) __bf16        v16bf;
typedef __attribute__((ext_vector_type(2)))  __bf16        v2bf;
typedef __attribute__((ext_vector_type(8)))  float         v8f;
typedef __attribute__((ext_vector_type(4)))  float         v4f;
typedef __attribute__((ext_vector_type(2)))  float         v2f;
typedef __attribute__((ext_vector_type(4)))  unsigned int  v4u;

union ABFrag { v4u q[2]; v16bf v; };
union CFrag  { v8f v; v4f h[2]; };

// Native bf16 convert (fptrunc, RTE) -> v_cvt_*bf16* on gfx1250
__device__ __forceinline__ unsigned pack2bf(v2f f) {
  v2bf b;
  b[0] = (__bf16)f[0];
  b[1] = (__bf16)f[1];
  return __builtin_bit_cast(unsigned, b);
}

__device__ __forceinline__ v8f wmma_bf16(const ABFrag& a, const ABFrag& b, v8f c) {
  return __builtin_amdgcn_wmma_f32_16x16x32_bf16(false, a.v, false, b.v,
                                                 (short)0, c, false, false);
}

// 16-bit A/B fragment per ISA 7.12.2: lane holds row (M or N) = lane&15,
// K chunks [8*half .. +7] and [16+8*half .. +7]  (two aligned 16B LDS loads).
__device__ __forceinline__ void load_frag(ABFrag& f, const unsigned short* base,
                                          int row, int stride, int kk, int half) {
  const unsigned short* p = base + row * stride + kk + 8 * half;
  f.q[0] = *(const v4u*)(p);
  f.q[1] = *(const v4u*)(p + 16);
}

// ============================ kernel 0: setup ============================
__global__ void k_setup(const float* __restrict__ omega,
                        const float* __restrict__ b_offset,
                        const float* __restrict__ tau,
                        float* __restrict__ ws, float* __restrict__ out) {
  int t = threadIdx.x;
  if (t < H_SZ) {
    float om = fabsf(omega[t]);
    ws[WS_OM + t] = om;
    float d = DT_C * om;
    float pom = (-1.0f + sqrtf(1.0f - d * d)) / DT_C;
    ws[WS_PB + t] = pom - fabsf(b_offset[t]);
  }
  if (t < OUT_SZ) ws[WS_AL + t] = expf(-DT_C / fabsf(tau[t]));
  if (t == 0) out[SS_OFF] = 0.0f;
}

// ================= kernel 1: cur_x = x @ W_x^T (WMMA bf16) =================
// Tile 128(M) x 64(N), K step 32. 21 guard-free K-steps + 1 tail (700=21*32+28).
// Staging: b64 f32-pair loads -> native bf16 pack -> ds_store_b32.
#define ASTR 40                       // ushorts (20 u32); 80B rows, 16B aligned
#define WSTR 712                      // ushorts (356 u32)
#define SMEM_B (128 * ASTR * 2 + 64 * WSTR * 2)   // 101,376 B

__device__ __forceinline__ void stage_A(unsigned* sA32, const float* __restrict__ x,
                                        int mBase, int kk, int tid) {
  const int k2 = tid & 15, rowBase = tid >> 4;       // pair index, 16 rows/pass
  #pragma unroll
  for (int p = 0; p < 8; ++p) {
    int row = rowBase + p * 16;
    v2f f = *(const v2f*)(x + (mBase + row) * IN_SZ + kk + 2 * k2);
    sA32[row * 20 + k2] = pack2bf(f);
  }
}

__global__ void k_gemm_x(const float* __restrict__ x,
                         const float* __restrict__ Wh,
                         float* __restrict__ curxP) {
  extern __shared__ v4u smem4[];
  char* smem = (char*)smem4;
  unsigned short* sA   = (unsigned short*)smem;                    // [128][ASTR]
  unsigned*       sA32 = (unsigned*)smem;
  unsigned short* sW   = (unsigned short*)(smem + 128 * ASTR * 2); // [64][WSTR]
  unsigned*       sW32 = (unsigned*)(smem + 128 * ASTR * 2);

  const int tid   = threadIdx.x;
  const int mBase = blockIdx.x * 128;
  const int nBase = blockIdx.y * 64;

  // Stage W panel once: 64 rows x 704 cols (352 pairs), zero-padded K
  for (int idx = tid; idx < 64 * 352; idx += 256) {
    int j = idx / 352, k2 = idx - j * 352, gk = 2 * k2;
    v2f f = {0.0f, 0.0f};
    if (gk < IN_SZ) f = *(const v2f*)(Wh + (nBase + j) * (IN_SZ + H_SZ) + gk);
    sW32[j * 356 + k2] = pack2bf(f);
  }

  const int wv = tid >> 5, lane = tid & 31, half = lane >> 4, ln = lane & 15;
  const int wm = (wv >> 1) * 32, wn = (wv & 1) * 32;

  v8f c00 = {}, c01 = {}, c10 = {}, c11 = {};

  for (int kk = 0; kk < K_PAD; kk += 32) {
    if (kk < IN_SZ - 28) {                               // 21 full tiles
      stage_A(sA32, x, mBase, kk, tid);
      __builtin_prefetch(x + (mBase + (tid >> 4)) * IN_SZ + kk + 32 + 2 * (tid & 15), 0, 1);
    } else {                                             // tail: k in [672,700)
      const int k2 = tid & 15, rowBase = tid >> 4;
      #pragma unroll
      for (int p = 0; p < 8; ++p) {
        int row = rowBase + p * 16;
        v2f f = {0.0f, 0.0f};
        if (k2 < 14) f = *(const v2f*)(x + (mBase + row) * IN_SZ + kk + 2 * k2);
        sA32[row * 20 + k2] = pack2bf(f);
      }
    }
    __syncthreads();

    ABFrag a0, a1, b0, b1;
    load_frag(a0, sA, wm + ln,      ASTR, 0,  half);
    load_frag(a1, sA, wm + 16 + ln, ASTR, 0,  half);
    load_frag(b0, sW, wn + ln,      WSTR, kk, half);
    load_frag(b1, sW, wn + 16 + ln, WSTR, kk, half);
    c00 = wmma_bf16(a0, b0, c00);
    c01 = wmma_bf16(a0, b1, c01);
    c10 = wmma_bf16(a1, b0, c10);
    c11 = wmma_bf16(a1, b1, c11);
    __syncthreads();
  }

  // Permuted store: curxP[(m>>4)*4096 + h*16 + (m&15)]; two b128 per fragment.
  const int g16a = (mBase + wm) >> 4;        // rows m .. m+15
  const int g16b = g16a + 1;                 // rows m+16 .. m+31
  const int n0 = nBase + wn + ln, n1 = n0 + 16;
  float* p00 = curxP + ((size_t)g16a * H_SZ + n0) * 16 + 8 * half;
  float* p01 = curxP + ((size_t)g16a * H_SZ + n1) * 16 + 8 * half;
  float* p10 = curxP + ((size_t)g16b * H_SZ + n0) * 16 + 8 * half;
  float* p11 = curxP + ((size_t)g16b * H_SZ + n1) * 16 + 8 * half;
  CFrag f;
  f.v = c00; *(v4f*)(p00) = f.h[0]; *(v4f*)(p00 + 4) = f.h[1];
  f.v = c01; *(v4f*)(p01) = f.h[0]; *(v4f*)(p01 + 4) = f.h[1];
  f.v = c10; *(v4f*)(p10) = f.h[0]; *(v4f*)(p10 + 4) = f.h[1];
  f.v = c11; *(v4f*)(p11) = f.h[0]; *(v4f*)(p11 + 4) = f.h[1];
}

// ============== kernel 2: persistent BRF recurrence + WMMA readout ==============
#define B_TILE 16
#define ZSTR   264                                   // 256+8 pad ushorts
#define C_WZ_OFF  0                                  // [256][ZSTR] bf16
#define C_Z_OFF   (H_SZ * ZSTR * 2)                  // 135,168: [16][ZSTR] bf16
#define C_WOB_OFF (C_Z_OFF + B_TILE * ZSTR * 2)      // 143,616: [32][ZSTR] bf16 (o padded)
#define C_AL_OFF  (C_WOB_OFF + 32 * ZSTR * 2)        // 160,512: [32] f32 (padded)
#define C_RED_OFF (C_AL_OFF + 32 * 4)                // 160,640: [256] f32
#define SMEM_C    (C_RED_OFF + 256 * 4)              // 161,664 B (< 320KB/WGP)

__global__ void k_recur(const float* __restrict__ Wh,
                        const float* __restrict__ Wout,
                        const float* __restrict__ ws,
                        const float* __restrict__ curxP,
                        float* __restrict__ out) {
  extern __shared__ v4u smem4[];
  char* smem = (char*)smem4;
  unsigned short* sWz  = (unsigned short*)(smem + C_WZ_OFF);
  unsigned short* sZ   = (unsigned short*)(smem + C_Z_OFF);
  unsigned short* sWoB = (unsigned short*)(smem + C_WOB_OFF);
  float* sAl  = (float*)(smem + C_AL_OFF);
  float* sRed = (float*)(smem + C_RED_OFF);

  const int tid = threadIdx.x;
  const int bt  = blockIdx.x;              // batch tile index
  const int bBase = bt * B_TILE;

  // Resident W_z (bf16) — pair-packed stores
  unsigned* sWz32 = (unsigned*)sWz;
  for (int idx = tid; idx < H_SZ * (H_SZ / 2); idx += 256) {
    int h = idx >> 7, k2 = idx & 127;
    v2f f = *(const v2f*)(Wh + h * (IN_SZ + H_SZ) + IN_SZ + 2 * k2);
    sWz32[h * 132 + k2] = pack2bf(f);
  }
  // W_out as bf16 B-fragments, o padded 20 -> 32 with zeros
  unsigned* sWoB32 = (unsigned*)sWoB;
  for (int idx = tid; idx < 32 * 128; idx += 256) {
    int o = idx >> 7, k2 = idx & 127;
    v2f f = {0.0f, 0.0f};
    if (o < OUT_SZ) f = *(const v2f*)(Wout + o * H_SZ + 2 * k2);
    sWoB32[o * 132 + k2] = pack2bf(f);
  }
  unsigned* sZ32 = (unsigned*)sZ;
  for (int idx = tid; idx < B_TILE * (ZSTR / 2); idx += 256) sZ32[idx] = 0;  // z0 = 0
  if (tid < 32) sAl[tid] = (tid < OUT_SZ) ? ws[WS_AL + tid] : 0.0f;

  const int wv = tid >> 5, lane = tid & 31, half = lane >> 4, ln = lane & 15;
  const int h0 = wv * 32 + ln, h1 = h0 + 16;
  const float om0 = ws[WS_OM + h0], om1 = ws[WS_OM + h1];
  const float pb0 = ws[WS_PB + h0], pb1 = ws[WS_PB + h1];

  float u0[8] = {}, v0[8] = {}, q0[8] = {}, z0[8] = {};
  float u1[8] = {}, v1[8] = {}, q1[8] = {}, z1[8] = {};
  float ou0[8] = {}, ou1[8] = {};          // out_u, valid in wave 0 only
  float spikeAcc = 0.0f;
  __syncthreads();
  const float a0v = sAl[ln], a1v = sAl[16 + ln];

  for (int t = 0; t < T_STEPS; ++t) {
    // Accumulator init from permuted cur_x: two b128 per fragment
    const float* cb = curxP + ((size_t)(t * 16 + bt) * H_SZ) * 16 + 8 * half;
    if (t + 1 < T_STEPS) __builtin_prefetch(cb + 16 * H_SZ * 16, 0, 1);
    CFrag f0, f1;
    f0.h[0] = *(const v4f*)(cb + h0 * 16);
    f0.h[1] = *(const v4f*)(cb + h0 * 16 + 4);
    f1.h[0] = *(const v4f*)(cb + h1 * 16);
    f1.h[1] = *(const v4f*)(cb + h1 * 16 + 4);
    v8f c0 = f0.v, c1 = f1.v;
    // cur += z_{t-1} @ W_z^T
    #pragma unroll
    for (int kk = 0; kk < H_SZ; kk += 32) {
      ABFrag a, b0f, b1f;
      load_frag(a,   sZ,  ln, ZSTR, kk, half);
      load_frag(b0f, sWz, h0, ZSTR, kk, half);
      load_frag(b1f, sWz, h1, ZSTR, kk, half);
      c0 = wmma_bf16(a, b0f, c0);
      c1 = wmma_bf16(a, b1f, c1);
    }
    // BRF neuron update (old u, old q semantics per reference)
    #pragma unroll
    for (int r = 0; r < 8; ++r) {
      {
        float bb = pb0 - q0[r];
        float un = u0[r] + DT_C * (bb * u0[r] - om0 * v0[r] + c0[r]);
        float vn = v0[r] + DT_C * (om0 * u0[r] + bb * v0[r]);
        float zz = (un - (THETA_C + q0[r]) > 0.0f) ? 1.0f : 0.0f;
        q0[r] = 0.9f * q0[r] + zz;
        u0[r] = un; v0[r] = vn; z0[r] = zz; spikeAcc += zz;
      }
      {
        float bb = pb1 - q1[r];
        float un = u1[r] + DT_C * (bb * u1[r] - om1 * v1[r] + c1[r]);
        float vn = v1[r] + DT_C * (om1 * u1[r] + bb * v1[r]);
        float zz = (un - (THETA_C + q1[r]) > 0.0f) ? 1.0f : 0.0f;
        q1[r] = 0.9f * q1[r] + zz;
        u1[r] = un; v1[r] = vn; z1[r] = zz; spikeAcc += zz;
      }
    }
    __syncthreads();   // all reads of sZ (GEMM + wave0 readout) done
    #pragma unroll
    for (int r = 0; r < 8; ++r) {
      int m = 8 * half + r;
      // spikes are exactly 0.0/1.0 -> store bf16 bit pattern directly
      sZ[m * ZSTR + h0] = (z0[r] != 0.0f) ? (unsigned short)0x3F80 : (unsigned short)0;
      sZ[m * ZSTR + h1] = (z1[r] != 0.0f) ? (unsigned short)0x3F80 : (unsigned short)0;
    }
    __syncthreads();   // z_t visible
    // WMMA readout (wave 0 only; EXEC all-ones inside the wave):
    // y[16x32] = z[16x256] @ WoB^T ; out_u = a*out_u + (1-a)*y ; stream outs[t]
    if (wv == 0) {
      v8f y0 = {}, y1 = {};
      #pragma unroll
      for (int kk = 0; kk < H_SZ; kk += 32) {
        ABFrag a, w0, w1;
        load_frag(a,  sZ,   ln,      ZSTR, kk, half);
        load_frag(w0, sWoB, ln,      ZSTR, kk, half);
        load_frag(w1, sWoB, 16 + ln, ZSTR, kk, half);
        y0 = wmma_bf16(a, w0, y0);
        y1 = wmma_bf16(a, w1, y1);
      }
      float* ob = out + OUTS_OFF + (size_t)t * (B_SZ * OUT_SZ);
      #pragma unroll
      for (int r = 0; r < 8; ++r) {
        int b = 8 * half + r;
        ou0[r] = a0v * ou0[r] + (1.0f - a0v) * y0[r];
        ob[(bBase + b) * OUT_SZ + ln] = ou0[r];
        ou1[r] = a1v * ou1[r] + (1.0f - a1v) * y1[r];
        if (ln < OUT_SZ - 16) ob[(bBase + b) * OUT_SZ + 16 + ln] = ou1[r];
      }
    }
  }

  // final states
  #pragma unroll
  for (int r = 0; r < 8; ++r) {
    int gb = bBase + 8 * half + r;
    out[FZ_OFF + gb * H_SZ + h0] = z0[r];
    out[FZ_OFF + gb * H_SZ + h1] = z1[r];
    out[FU_OFF + gb * H_SZ + h0] = u0[r];
    out[FU_OFF + gb * H_SZ + h1] = u1[r];
  }
  if (wv == 0) {
    #pragma unroll
    for (int r = 0; r < 8; ++r) {
      int gb = bBase + 8 * half + r;
      out[FOUT_OFF + gb * OUT_SZ + ln] = ou0[r];
      if (ln < OUT_SZ - 16) out[FOUT_OFF + gb * OUT_SZ + 16 + ln] = ou1[r];
    }
  }
  sRed[tid] = spikeAcc;
  __syncthreads();
  if (tid == 0) {
    float s = 0.0f;
    for (int i = 0; i < 256; ++i) s += sRed[i];
    atomicAdd(&out[SS_OFF], s);
  }
}

// ================================ launch ================================
extern "C" void kernel_launch(void* const* d_in, const int* in_sizes, int n_in,
                              void* d_out, int out_size, void* d_ws, size_t ws_size,
                              hipStream_t stream) {
  (void)in_sizes; (void)n_in; (void)out_size; (void)ws_size;
  const float* x    = (const float*)d_in[0];   // [250,256,700]
  const float* Wh   = (const float*)d_in[1];   // [256,956]
  const float* omg  = (const float*)d_in[2];   // [256]
  const float* bof  = (const float*)d_in[3];   // [256]
  const float* Wo   = (const float*)d_in[4];   // [20,256]
  const float* tau  = (const float*)d_in[5];   // [20]
  float* out  = (float*)d_out;
  float* ws   = (float*)d_ws;
  float* curxP = ws + WS_CURX;                 // ~65.5 MB scratch

  k_setup<<<1, 256, 0, stream>>>(omg, bof, tau, ws, out);

  dim3 gB(M_ROWS / 128, H_SZ / 64);            // 500 x 4 tiles
  k_gemm_x<<<gB, 256, SMEM_B, stream>>>(x, Wh, curxP);

  k_recur<<<B_SZ / B_TILE, 256, SMEM_C, stream>>>(Wh, Wo, ws, curxP, out);
}
